// CausalTransformer_6253472383390
// MI455X (gfx1250) — matrix operand order repaired and checked by reading, not compiled, not run on hardware
//
#include <hip/hip_runtime.h>
#include <stdint.h>

// ---------------------------------------------------------------------------
// Problem dims (match reference)
// ---------------------------------------------------------------------------
static constexpr int kTOK = 8192;   // N tokens
static constexpr int kHID = 2048;
static constexpr int kFF  = 8192;   // 4*HID
static constexpr int kNH  = 16;     // heads
static constexpr int kDH  = 128;    // head dim

// ---------------------------------------------------------------------------
// WMMA types (CDNA5 gfx1250, wave32)
// ---------------------------------------------------------------------------
typedef __bf16 bf16_t;
typedef bf16_t v16bf __attribute__((ext_vector_type(16)));
typedef float  v8f   __attribute__((ext_vector_type(8)));

// async-copy pointer types: builtin expects int4-vector pointers
// (diagnostic: "__attribute__((__vector_size__(4 * sizeof(int)))) int __device__ *")
typedef int v4i_vs __attribute__((vector_size(16)));
typedef __attribute__((address_space(1))) v4i_vs* glb_v4i_p;
typedef __attribute__((address_space(3))) v4i_vs* lds_v4i_p;

#if __has_builtin(__builtin_amdgcn_global_load_async_to_lds_b128)
#define ASYNC_COPY 1
#else
#define ASYNC_COPY 0
#endif

#if __has_builtin(__builtin_amdgcn_s_wait_asynccnt)
#define WAIT_ASYNC(n) __builtin_amdgcn_s_wait_asynccnt(n)
#else
#define WAIT_ASYNC(n) asm volatile("s_wait_asynccnt %0" ::"i"(n) : "memory")
#endif

__device__ __forceinline__ uint16_t f2bf(float f) {
  uint32_t u = __float_as_uint(f);
  uint32_t r = (u + 0x7FFFu + ((u >> 16) & 1u)) >> 16;  // round-to-nearest-even
  return (uint16_t)r;
}

// ---------------------------------------------------------------------------
// fp32 -> bf16 converter (activations)
// ---------------------------------------------------------------------------
__global__ void cvt_f32_bf16_kernel(const float* __restrict__ x,
                                    uint16_t* __restrict__ y, int n) {
  int i = blockIdx.x * blockDim.x + threadIdx.x;
  int stride = gridDim.x * blockDim.x;
  for (; i < n; i += stride) y[i] = f2bf(x[i]);
}

// (a - b) -> bf16
__global__ void sub_f32_to_bf16_kernel(const float* __restrict__ a,
                                       const float* __restrict__ b,
                                       uint16_t* __restrict__ y, int n) {
  int i = blockIdx.x * blockDim.x + threadIdx.x;
  int stride = gridDim.x * blockDim.x;
  for (; i < n; i += stride) y[i] = f2bf(a[i] - b[i]);
}

// ---------------------------------------------------------------------------
// Weight pre-swizzle: fp32 W[K][N] -> bf16-pair tiled Wp[K/32][N][16] dwords.
//   dword (kb, n, k2l) = { W[2*(kb*16+k2l)][n] , W[2*(kb*16+k2l)+1][n] }
// This matches the per-lane B-fragment layout of v_wmma_*_16x16x32_bf16, so
// the GEMM's B tile staging is a straight async copy and the B fragment read
// is two ds_load_b128.
// ---------------------------------------------------------------------------
__global__ void pack_weight_kernel(const float* __restrict__ W,
                                   uint32_t* __restrict__ out, int N) {
  int n = blockIdx.x * blockDim.x + threadIdx.x;  // column
  int k2 = blockIdx.y;                            // K-pair index
  uint32_t lo = f2bf(W[(size_t)(2 * k2) * N + n]);
  uint32_t hi = f2bf(W[(size_t)(2 * k2 + 1) * N + n]);
  out[((size_t)(k2 >> 4) * N + n) * 16 + (k2 & 15)] = lo | (hi << 16);
}

// ---------------------------------------------------------------------------
// bf16 GEMM:  C[kTOK,ND] = act(A[kTOK,KD] @ W + bias)
//   A row-major bf16; W pre-swizzled (above); fp32 accumulate via
//   v_wmma_f32_16x16x32_bf16.  Block tile 128x128, BLOCK_K=32, 256 thr =
//   8 waves (4M x 2N), wave tile 32x64 = 2x4 WMMA tiles.  Double-buffered
//   LDS staged with global_load_async_to_lds_b128 (ASYNCcnt) when available.
// ACT: 0 none, 1 relu, 2 leaky(0.01).  WBF/WF: write bf16 / f32 outputs.
// ---------------------------------------------------------------------------
template <int ACT, bool WBF, bool WF, int KD, int ND>
__global__ __launch_bounds__(256)
void gemm_bf16_wmma(const uint16_t* __restrict__ A,
                    const uint32_t* __restrict__ Bp,
                    const float* __restrict__ bias,
                    uint16_t* __restrict__ outBf,
                    float* __restrict__ outF) {
  constexpr int TILES = KD / 32;
  // rows padded 16 -> 20 dwords (80 B: 16B-aligned, bank-conflict-free)
  __shared__ uint32_t As[2][128 * 20];
  __shared__ uint32_t Bs[2][128 * 20];

  const int tid = threadIdx.x;
  const int lane = tid & 31;
  const int wave = tid >> 5;
  const int waveM = wave >> 1;  // 0..3
  const int waveN = wave & 1;   // 0..1
  const int blockN0 = blockIdx.x * 128;
  const int blockM0 = blockIdx.y * 128;
  const int lhalf = lane >> 4;  // 0: lanes 0-15, 1: lanes 16-31
  const int l15 = lane & 15;

  // stage one 32-K tile (A: 128x32 bf16, B: 128 cols x 16 dwords) = 512+512
  // 16B chunks; 2 A-chunks + 2 B-chunks per thread => 4 async ops / wave.
  auto issue_tile = [&](int t, int buf) {
    const int k0 = t * 32;
#pragma unroll
    for (int it = 0; it < 2; ++it) {
      int c = tid + it * 256;
      int row = c >> 2, seg = c & 3;  // row = A-row or B-column (0..127)
      const uint16_t* gA = A + (size_t)(blockM0 + row) * KD + k0 + seg * 8;
      uint32_t* lA = &As[buf][row * 20 + seg * 4];
      const uint32_t* gB = Bp + ((size_t)t * ND + (blockN0 + row)) * 16 + seg * 4;
      uint32_t* lB = &Bs[buf][row * 20 + seg * 4];
#if ASYNC_COPY
      __builtin_amdgcn_global_load_async_to_lds_b128(
          (glb_v4i_p)(uintptr_t)gA, (lds_v4i_p)(uintptr_t)lA, 0, 0);
      __builtin_amdgcn_global_load_async_to_lds_b128(
          (glb_v4i_p)(uintptr_t)gB, (lds_v4i_p)(uintptr_t)lB, 0, 0);
#else
      *reinterpret_cast<uint4*>(lA) = *reinterpret_cast<const uint4*>(gA);
      *reinterpret_cast<uint4*>(lB) = *reinterpret_cast<const uint4*>(gB);
#endif
    }
  };

  v8f acc[2][4];
  const v8f vzero = {0.f, 0.f, 0.f, 0.f, 0.f, 0.f, 0.f, 0.f};
#pragma unroll
  for (int i = 0; i < 2; ++i)
#pragma unroll
    for (int j = 0; j < 4; ++j) acc[i][j] = vzero;

  union Frag { v16bf v; uint32_t u[8]; uint4 q[2]; };

  issue_tile(0, 0);

  for (int t = 0; t < TILES; ++t) {
    const int buf = t & 1;
    if (t + 1 < TILES) {
      issue_tile(t + 1, buf ^ 1);  // prefetch next tile into other buffer
#if ASYNC_COPY
      WAIT_ASYNC(4);               // current tile's 4 ops (per wave) done
#endif
    } else {
#if ASYNC_COPY
      WAIT_ASYNC(0);
#endif
    }
    __syncthreads();  // all waves' async writes for tile t visible

    // A fragments: lanes 0-15 K{0-7,16-23}, lanes 16-31 K{8-15,24-31}
    Frag af[2];
#pragma unroll
    for (int i = 0; i < 2; ++i) {
      int m = waveM * 32 + i * 16 + l15;
      int base = m * 20 + lhalf * 4;
      af[i].q[0] = *reinterpret_cast<const uint4*>(&As[buf][base]);
      af[i].q[1] = *reinterpret_cast<const uint4*>(&As[buf][base + 8]);
    }
    // B fragments: lanes 0-15 K{0-7,16-23}, lanes 16-31 K{8-15,24-31}
    Frag bfr[4];
#pragma unroll
    for (int j = 0; j < 4; ++j) {
      int col = waveN * 64 + j * 16 + l15;
      int base = col * 20 + lhalf * 4;
      bfr[j].q[0] = *reinterpret_cast<const uint4*>(&Bs[buf][base]);
      bfr[j].q[1] = *reinterpret_cast<const uint4*>(&Bs[buf][base + 8]);
    }
#pragma unroll
    for (int i = 0; i < 2; ++i)
#pragma unroll
      for (int j = 0; j < 4; ++j)
        acc[i][j] = __builtin_amdgcn_wmma_f32_16x16x32_bf16(
            false, af[i].v, false, bfr[j].v, (short)0, acc[i][j], false, false);

    __syncthreads();  // tile t fully consumed before buf is re-filled
  }

  // ---- epilogue: bias + activation; compile-time strides, no branches ----
  // C layout: VGPR r -> row r (lanes 0-15) / 8+r (lanes 16-31), col = lane&15
#pragma unroll
  for (int i = 0; i < 2; ++i) {
    int gm = blockM0 + waveM * 32 + i * 16 + lhalf * 8;
#pragma unroll
    for (int j = 0; j < 4; ++j) {
      int gn = blockN0 + waveN * 64 + j * 16 + l15;
      float bv = bias[gn];
#pragma unroll
      for (int r = 0; r < 8; ++r) {
        float v = acc[i][j][r] + bv;
        if (ACT == 1) v = fmaxf(v, 0.f);
        if (ACT == 2) v = (v >= 0.f) ? v : 0.01f * v;
        size_t idx = (size_t)(gm + r) * ND + gn;
        if (WBF) outBf[idx] = f2bf(v);
        if (WF) outF[idx] = v;
      }
    }
  }
}

// ---------------------------------------------------------------------------
// Per-token residual-adjusted attention (one block per token).
// scores[h][g] = Q[h]·K[g]/sqrt(DH) - Q[h]·R[g]; softmax over g; out = attn@V.
// ~1.6 GFLOP total -> VALU; WMMA reserved for the big GEMMs.
// ---------------------------------------------------------------------------
__global__ __launch_bounds__(256)
void attn_kernel(const float* __restrict__ Q, const float* __restrict__ K,
                 const float* __restrict__ V, const float* __restrict__ R,
                 uint16_t* __restrict__ AO) {
  __shared__ float sQ[kNH][kDH];
  __shared__ float sK[kNH][kDH];
  __shared__ float sV[kNH][kDH];
  __shared__ float sR[kNH][kDH];
  __shared__ float sA[kNH][kNH];

  const int n = blockIdx.x;
  const int tid = threadIdx.x;
  const size_t base = (size_t)n * kHID;

  for (int i = tid; i < kHID; i += 256) {
    int h = i >> 7, d = i & 127;
    sQ[h][d] = Q[base + i];
    sK[h][d] = K[base + i];
    sV[h][d] = V[base + i];
    sR[h][d] = R[base + i];
  }
  __syncthreads();

  {
    int h = tid >> 4, g = tid & 15;
    float s = 0.f, s2 = 0.f;
#pragma unroll 8
    for (int d = 0; d < kDH; ++d) {
      float q = sQ[h][d];
      s += q * sK[g][d];
      s2 += q * sR[g][d];
    }
    float score = s * 0.0883883476483184405f - s2;  // 1/sqrt(128); Q·R unscaled
    float m = score;
#pragma unroll
    for (int off = 8; off >= 1; off >>= 1) m = fmaxf(m, __shfl_xor(m, off, 16));
    float e = expf(score - m);
    float sum = e;
#pragma unroll
    for (int off = 8; off >= 1; off >>= 1) sum += __shfl_xor(sum, off, 16);
    sA[h][g] = e / sum;
  }
  __syncthreads();

  {
    int h = tid >> 4;
    int dbase = (tid & 15) * 8;
#pragma unroll
    for (int j = 0; j < 8; ++j) {
      int d = dbase + j;
      float o = 0.f;
#pragma unroll
      for (int g = 0; g < 16; ++g) o += sA[h][g] * sV[g][d];
      AO[base + h * kDH + d] = f2bf(o);
    }
  }
}

// ---------------------------------------------------------------------------
// y = LayerNorm(X + A) * g + b ; one block per row; fp32 out + optional bf16
// ---------------------------------------------------------------------------
__global__ __launch_bounds__(256)
void add_ln_kernel(const float* __restrict__ X, const float* __restrict__ A,
                   const float* __restrict__ g, const float* __restrict__ b,
                   float* __restrict__ outF, uint16_t* __restrict__ outBf) {
  __shared__ float sv[kHID];
  __shared__ float red[256];
  __shared__ float red2[256];
  const int row = blockIdx.x, tid = threadIdx.x;
  const size_t base = (size_t)row * kHID;

  float s = 0.f, s2 = 0.f;
  for (int i = tid; i < kHID; i += 256) {
    float v = X[base + i] + A[base + i];
    sv[i] = v;
    s += v;
    s2 += v * v;
  }
  red[tid] = s;
  red2[tid] = s2;
  __syncthreads();
  for (int off = 128; off > 0; off >>= 1) {
    if (tid < off) {
      red[tid] += red[tid + off];
      red2[tid] += red2[tid + off];
    }
    __syncthreads();
  }
  const float mean = red[0] * (1.f / kHID);
  const float var = red2[0] * (1.f / kHID) - mean * mean;
  const float rstd = rsqrtf(var + 1e-5f);
  for (int i = tid; i < kHID; i += 256) {
    float y = (sv[i] - mean) * rstd * g[i] + b[i];
    outF[base + i] = y;
    if (outBf) outBf[base + i] = f2bf(y);
  }
}

// ---------------------------------------------------------------------------
// Host orchestration
// ---------------------------------------------------------------------------
extern "C" void kernel_launch(void* const* d_in, const int* in_sizes, int n_in,
                              void* d_out, int out_size, void* d_ws, size_t ws_size,
                              hipStream_t stream) {
  (void)in_sizes; (void)n_in; (void)out_size; (void)ws_size;
  const float* emb = (const float*)d_in[0];
  const float* obs = (const float*)d_in[1];
  const float* pw1 = (const float*)d_in[2];  const float* pb1 = (const float*)d_in[3];
  const float* pw2 = (const float*)d_in[4];  const float* pb2 = (const float*)d_in[5];
  const float* ow1 = (const float*)d_in[6];  const float* ob1 = (const float*)d_in[7];
  const float* ow2 = (const float*)d_in[8];  const float* ob2 = (const float*)d_in[9];
  const float* rw1 = (const float*)d_in[10]; const float* rb1 = (const float*)d_in[11];
  const float* rw2 = (const float*)d_in[12]; const float* rb2 = (const float*)d_in[13];
  const float* wq  = (const float*)d_in[14]; const float* bq  = (const float*)d_in[15];
  const float* wk  = (const float*)d_in[16]; const float* bk  = (const float*)d_in[17];
  const float* wv  = (const float*)d_in[18]; const float* bv  = (const float*)d_in[19];
  const float* wo  = (const float*)d_in[20]; const float* bo  = (const float*)d_in[21];
  const float* fw1 = (const float*)d_in[22]; const float* fb1 = (const float*)d_in[23];
  const float* fw2 = (const float*)d_in[24]; const float* fb2 = (const float*)d_in[25];
  const float* g1  = (const float*)d_in[26]; const float* be1 = (const float*)d_in[27];
  const float* g2  = (const float*)d_in[28]; const float* be2 = (const float*)d_in[29];

  const size_t NHID = (size_t)kTOK * kHID;
  const size_t H2   = (size_t)kHID * kHID;
  const size_t HF   = (size_t)kHID * kFF;

  char* p = (char*)d_ws;
  auto carve = [&](size_t bytes) -> void* {
    void* r = (void*)p;
    p += (bytes + 255) & ~(size_t)255;
    return r;
  };

  // pre-swizzled bf16-pair weights (K*N/2 dwords each == K*N bf16 bytes)
  uint32_t* w_pw1 = (uint32_t*)carve(H2 * 2);
  uint32_t* w_pw2 = (uint32_t*)carve(H2 * 2);
  uint32_t* w_ow1 = (uint32_t*)carve(H2 * 2);
  uint32_t* w_ow2 = (uint32_t*)carve(H2 * 2);
  uint32_t* w_rw1 = (uint32_t*)carve(H2 * 2);
  uint32_t* w_rw2 = (uint32_t*)carve(H2 * 2);
  uint32_t* w_wq  = (uint32_t*)carve(H2 * 2);
  uint32_t* w_wk  = (uint32_t*)carve(H2 * 2);
  uint32_t* w_wv  = (uint32_t*)carve(H2 * 2);
  uint32_t* w_wo  = (uint32_t*)carve(H2 * 2);
  uint32_t* w_fw1 = (uint32_t*)carve(HF * 2);
  uint32_t* w_fw2 = (uint32_t*)carve(HF * 2);
  // bf16 activations
  uint16_t* ebf   = (uint16_t*)carve(NHID * 2);
  uint16_t* obf   = (uint16_t*)carve(NHID * 2);
  uint16_t* hbf   = (uint16_t*)carve((size_t)kTOK * kFF * 2);  // FFN hidden
  uint16_t* dbf   = (uint16_t*)carve(NHID * 2);
  uint16_t* aobf  = (uint16_t*)carve(NHID * 2);
  uint16_t* attbf = (uint16_t*)carve(NHID * 2);
  // fp32 activation buffers (rotated)
  float* f1 = (float*)carve(NHID * 4);
  float* f2 = (float*)carve(NHID * 4);
  float* f3 = (float*)carve(NHID * 4);
  float* f4 = (float*)carve(NHID * 4);

  auto pack = [&](const float* src, uint32_t* dst, int K, int N) {
    pack_weight_kernel<<<dim3(N / 256, K / 2), 256, 0, stream>>>(src, dst, N);
  };
  auto cvt = [&](const float* src, uint16_t* dst, size_t n) {
    cvt_f32_bf16_kernel<<<2048, 256, 0, stream>>>(src, dst, (int)n);
  };

  pack(pw1, w_pw1, kHID, kHID); pack(pw2, w_pw2, kHID, kHID);
  pack(ow1, w_ow1, kHID, kHID); pack(ow2, w_ow2, kHID, kHID);
  pack(rw1, w_rw1, kHID, kHID); pack(rw2, w_rw2, kHID, kHID);
  pack(wq,  w_wq,  kHID, kHID); pack(wk,  w_wk,  kHID, kHID);
  pack(wv,  w_wv,  kHID, kHID); pack(wo,  w_wo,  kHID, kHID);
  pack(fw1, w_fw1, kHID, kFF);  pack(fw2, w_fw2, kFF, kHID);
  cvt(emb, ebf, NHID);
  cvt(obs, obf, NHID);

  const dim3 blk(256);
  const dim3 gHH(kHID / 128, kTOK / 128);   // 2048-col GEMMs
  const dim3 gHF(kFF / 128, kTOK / 128);    // 8192-col GEMM

  // hidden = relu(X @ W1 + b1)  -> bf16
  auto gemm_h = [&](const uint16_t* A, const uint32_t* W, const float* bias, uint16_t* oB) {
    gemm_bf16_wmma<1, true, false, kHID, kHID><<<gHH, blk, 0, stream>>>(A, W, bias, oB, nullptr);
  };
  // out = X @ W2 + b2 -> f32
  auto gemm_o = [&](const uint16_t* A, const uint32_t* W, const float* bias, float* oF) {
    gemm_bf16_wmma<0, false, true, kHID, kHID><<<gHH, blk, 0, stream>>>(A, W, bias, nullptr, oF);
  };

  // predictor / observer / residual MLPs
  gemm_h(ebf, w_pw1, pb1, hbf);
  gemm_o(hbf, w_pw2, pb2, f1);                                        // P
  gemm_h(obf, w_ow1, ob1, hbf);
  gemm_o(hbf, w_ow2, ob2, f2);                                        // O
  sub_f32_to_bf16_kernel<<<2048, 256, 0, stream>>>(f1, f2, dbf, (int)NHID);
  gemm_h(dbf, w_rw1, rb1, hbf);
  gemm_o(hbf, w_rw2, rb2, f3);                                        // R

  // Q/K/V projections
  gemm_o(ebf, w_wq, bq, f1);                                          // Q
  gemm_o(ebf, w_wk, bk, f2);                                          // K
  gemm_o(ebf, w_wv, bv, f4);                                          // V

  attn_kernel<<<kTOK, 256, 0, stream>>>(f1, f2, f4, f3, aobf);

  gemm_o(aobf, w_wo, bo, f3);                                         // attended
  add_ln_kernel<<<kTOK, 256, 0, stream>>>(emb, f3, g1, be1, f1, attbf);  // ATT

  // FFN: leaky(0.01) expand to 8192, then contract
  gemm_bf16_wmma<2, true, false, kHID, kFF><<<gHF, blk, 0, stream>>>(attbf, w_fw1, fb1, hbf, nullptr);
  gemm_bf16_wmma<0, false, true, kFF, kHID><<<gHH, blk, 0, stream>>>(hbf, w_fw2, fb2, nullptr, f2);  // F

  add_ln_kernel<<<kTOK, 256, 0, stream>>>(f1, f2, g2, be2, (float*)d_out, nullptr);
}
